// HeteroGNN_LSTM_49752901157181
// MI455X (gfx1250) — matrix operand order repaired and checked by reading, not compiled, not run on hardware
//
#include <hip/hip_runtime.h>
#include <hip/hip_bf16.h>

// ---------------------------------------------------------------------------
// Types for CDNA5 WMMA
// ---------------------------------------------------------------------------
typedef __attribute__((ext_vector_type(16))) __bf16 v16bf;
typedef __attribute__((ext_vector_type(8)))  float  v8f;

union BFrag {            // one WMMA B fragment per lane: 8 dwords = 16 bf16
    uint4 q[2];
    v16bf v;
};

__device__ __forceinline__ float sigf(float x) {
    return 1.0f / (1.0f + __expf(-x));
}

// ---------------------------------------------------------------------------
// Sizes (compile-time constants from the reference)
// ---------------------------------------------------------------------------
#define N_HRU 8192
#define N_GW  150000
#define N_CH  150000
#define E_SWGW 400000
#define E_SW   400000
#define D_IN  128
#define D_H   256
#define D_G   1024   // 4*D_H
#define COLT  4      // 16x16 col tiles per wave (wave computes 16x64 strip)

// ---------------------------------------------------------------------------
// Init: zero degree counters, seed out_hru with (b_swgw + b_sw)
// ---------------------------------------------------------------------------
__global__ void k_init(float* __restrict__ out_hru,
                       const float* __restrict__ b1, const float* __restrict__ b2,
                       unsigned* __restrict__ dAs, unsigned* __restrict__ dAd,
                       unsigned* __restrict__ dBs, unsigned* __restrict__ dBd) {
    int i = blockIdx.x * blockDim.x + threadIdx.x;
    if (i < N_HRU * D_H) out_hru[i] = b1[i & (D_H - 1)] + b2[i & (D_H - 1)];
    if (i < N_GW)  { dAs[i] = 0u; dBs[i] = 0u; }   // N_GW == N_CH
    if (i < N_HRU) { dAd[i] = 0u; dBd[i] = 0u; }
}

// ---------------------------------------------------------------------------
// Degree counting (segment_sum of ones over src and dst)
// ---------------------------------------------------------------------------
__global__ void k_deg(const int* __restrict__ src, const int* __restrict__ dst,
                      unsigned* __restrict__ dS, unsigned* __restrict__ dD, int E) {
    int i = blockIdx.x * blockDim.x + threadIdx.x;
    if (i < E) {
        atomicAdd(&dS[src[i]], 1u);
        atomicAdd(&dD[dst[i]], 1u);
    }
}

// ---------------------------------------------------------------------------
// Pre-pack a weight matrix into bf16 WMMA B-fragment layout:
//   Bp[((kb*(N/16) + nt)*32 + lane)*8 + j] = pack_bf16(B(k,col), B(k+1,col))
// with col = nt*16 + (lane&15), k = kb*32 + (j>>2)*16 + (lane>>4)*8 + 2*(j&3)
// (ISA 7.12.2 B 32x16 layout: lane = column; lanes 0-15 K=0..7,16..23, 16-31 K=8..15,24..31).
// trans=0: B(k,n) = W[k*N+n] (row-major [K,N]);  trans=1: B(k,n) = W[n*K+k].
// ---------------------------------------------------------------------------
__global__ void k_pack(const float* __restrict__ W, unsigned* __restrict__ Bp,
                       int K, int N, int trans) {
    const int i = blockIdx.x * blockDim.x + threadIdx.x;
    const int total = (K / 32) * (N / 16) * 256;
    if (i >= total) return;
    const int j    = i & 7;
    const int lane = (i >> 3) & 31;
    const int rest = i >> 8;
    const int ntiles = N / 16;
    const int nt = rest % ntiles;
    const int kb = rest / ntiles;
    const int col = nt * 16 + (lane & 15);
    const int k   = kb * 32 + (j >> 2) * 16 + (lane >> 4) * 8 + 2 * (j & 3);
    const float f0 = trans ? W[(long)col * K + k]       : W[(long)k * N + col];
    const float f1 = trans ? W[(long)col * K + k + 1]   : W[(long)(k + 1) * N + col];
    union { __bf16 h[2]; unsigned u; } pk;
    pk.h[0] = (__bf16)f0;     // low 16 bits = even K  (bv[2j])
    pk.h[1] = (__bf16)f1;     // high 16 bits = odd K  (bv[2j+1])
    Bp[i] = pk.u;
}

// ---------------------------------------------------------------------------
// Load the per-lane A fragment (16x32 bf16, ISA 7.12.2 layout) for k-block kk.
// Per lane the needed K values are two contiguous 8-float runs -> 4 b128 loads,
// converted with packed v_cvt_pk_bf16_f32.
// ---------------------------------------------------------------------------
__device__ __forceinline__ v16bf load_a_frag(const float* __restrict__ arow,
                                             int kk, int kh) {
    const float4 c0 = *(const float4*)(arow + kk + kh * 8);
    const float4 c1 = *(const float4*)(arow + kk + kh * 8 + 4);
    const float4 c2 = *(const float4*)(arow + kk + 16 + kh * 8);
    const float4 c3 = *(const float4*)(arow + kk + 16 + kh * 8 + 4);
    v16bf av;
    av[0]  = (__bf16)c0.x; av[1]  = (__bf16)c0.y; av[2]  = (__bf16)c0.z; av[3]  = (__bf16)c0.w;
    av[4]  = (__bf16)c1.x; av[5]  = (__bf16)c1.y; av[6]  = (__bf16)c1.z; av[7]  = (__bf16)c1.w;
    av[8]  = (__bf16)c2.x; av[9]  = (__bf16)c2.y; av[10] = (__bf16)c2.z; av[11] = (__bf16)c2.w;
    av[12] = (__bf16)c3.x; av[13] = (__bf16)c3.y; av[14] = (__bf16)c3.z; av[15] = (__bf16)c3.w;
    return av;
}

// ---------------------------------------------------------------------------
// Unified WMMA GEMM against pre-packed B:
//   C[row,col] = sum_k A[row,k] * B(k,col) (+ bias1[col] + bias2[col])
// grid = (rows/16, N/(16*COLT)), block = 32.  Wave owns a 16x64 output strip;
// A fragment loaded once per k-block, B fragments are two b128 loads each.
// ---------------------------------------------------------------------------
__global__ void k_gemm(const float* __restrict__ A, const unsigned* __restrict__ Bp,
                       const float* __restrict__ bias1, const float* __restrict__ bias2,
                       float* __restrict__ C, int K, int N) {
    const int l  = threadIdx.x;
    const int m  = l & 15;      // A row in tile / C column in tile
    const int kh = l >> 4;      // lane half selects K sub-range
    const long row = (long)blockIdx.x * 16 + m;
    const int  nt0 = blockIdx.y * COLT;
    const int  ntiles = N / 16;
    const float* arow = A + row * K;

    v8f acc[COLT];
    #pragma unroll
    for (int ct = 0; ct < COLT; ++ct) acc[ct] = (v8f){};

    for (int kb = 0; kb < K / 32; ++kb) {
        const v16bf av = load_a_frag(arow, kb * 32, kh);
        #pragma unroll
        for (int ct = 0; ct < COLT; ++ct) {
            const uint4* bp = (const uint4*)(Bp +
                (((long)(kb * ntiles + nt0 + ct)) * 32 + l) * 8);
            BFrag b;
            b.q[0] = bp[0];
            b.q[1] = bp[1];
            acc[ct] = __builtin_amdgcn_wmma_f32_16x16x32_bf16(
                false, av, false, b.v, (short)0, acc[ct], false, false);
        }
    }
    // C/D f32 layout: VGPR j -> row j + 8*kh, column = lane&15
    const long rb = (long)blockIdx.x * 16;
    #pragma unroll
    for (int ct = 0; ct < COLT; ++ct) {
        const int col = (nt0 + ct) * 16 + m;
        const float bb = (bias1 ? bias1[col] : 0.0f) + (bias2 ? bias2[col] : 0.0f);
        #pragma unroll
        for (int j = 0; j < 8; ++j)
            C[(rb + j + 8 * kh) * N + col] = acc[ct][j] + bb;
    }
}

// ---------------------------------------------------------------------------
// Edge scatter: out[dst] += norm_e * h[src], norm_e = invsqrt(degS)*invsqrt(degD)
// One 256-thread block per edge (one thread per hidden dim).
// ---------------------------------------------------------------------------
__global__ void k_scatter(const float* __restrict__ h,
                          const int* __restrict__ src, const int* __restrict__ dst,
                          const unsigned* __restrict__ degS, const unsigned* __restrict__ degD,
                          float* __restrict__ out) {
    __shared__ float nrm;
    __shared__ int ss, dd;
    const int e = blockIdx.x;
    if (threadIdx.x == 0) {
        int s = src[e], d = dst[e];
        unsigned a = degS[s], b = degD[d];
        float ia = a ? rsqrtf((float)a) : 0.0f;
        float ib = b ? rsqrtf((float)b) : 0.0f;
        ss = s; dd = d; nrm = ia * ib;
    }
    __syncthreads();
    const int t = threadIdx.x;
    atomicAdd(&out[(long)dd * D_H + t], nrm * h[(long)ss * D_H + t]);
}

// ---------------------------------------------------------------------------
// Sequential LSTM layer. One 1024-thread block; h/c/gates in LDS.
// xg[t,j] already contains x@Wih.T + bih + bhh.  PyTorch gate order i,f,g,o.
// ---------------------------------------------------------------------------
__global__ void k_lstm(const float* __restrict__ xg, const float* __restrict__ Whh,
                       float* __restrict__ hseq, int T) {
    __shared__ float hsh[D_H];
    __shared__ float csh[D_H];
    __shared__ float gsh[D_G];
    const int j = threadIdx.x;
    if (j < D_H) { hsh[j] = 0.0f; csh[j] = 0.0f; }
    __syncthreads();

    const float4* w4 = (const float4*)(Whh + (long)j * D_H);
    for (int t = 0; t < T; ++t) {
        float acc = xg[(long)t * D_G + j];
        const float4* h4 = (const float4*)hsh;
        #pragma unroll 8
        for (int k = 0; k < D_H / 4; ++k) {
            float4 w = w4[k];
            float4 h = h4[k];
            acc += w.x * h.x + w.y * h.y + w.z * h.z + w.w * h.w;
        }
        gsh[j] = acc;
        __syncthreads();
        if (j < D_H) {
            float ig = sigf(gsh[j]);
            float fg = sigf(gsh[D_H + j]);
            float gc = tanhf(gsh[2 * D_H + j]);
            float og = sigf(gsh[3 * D_H + j]);
            float cn = fg * csh[j] + ig * gc;
            csh[j] = cn;
            float hn = og * tanhf(cn);
            hsh[j] = hn;
            hseq[(long)t * D_H + j] = hn;
        }
        __syncthreads();
    }
}

// ---------------------------------------------------------------------------
// FC head: out = relu(h_last @ fc1 + b1) @ fc2 + b2   (scalar)
// ---------------------------------------------------------------------------
__global__ void k_head(const float* __restrict__ hlast,
                       const float* __restrict__ fc1W, const float* __restrict__ fc1b,
                       const float* __restrict__ fc2W, const float* __restrict__ fc2b,
                       float* __restrict__ out) {
    __shared__ float red[128];
    const int j = threadIdx.x;
    float acc = fc1b[j];
    #pragma unroll 4
    for (int k = 0; k < D_H; ++k) acc += hlast[k] * fc1W[k * 128 + j];
    red[j] = fmaxf(acc, 0.0f) * fc2W[j];
    __syncthreads();
    for (int s = 64; s > 0; s >>= 1) {
        if (j < s) red[j] += red[j + s];
        __syncthreads();
    }
    if (j == 0) out[0] = red[0] + fc2b[0];
}

// ---------------------------------------------------------------------------
// Launch
// ---------------------------------------------------------------------------
extern "C" void kernel_launch(void* const* d_in, const int* in_sizes, int n_in,
                              void* d_out, int out_size, void* d_ws, size_t ws_size,
                              hipStream_t stream) {
    (void)in_sizes; (void)n_in; (void)out_size; (void)ws_size;

    const float* x_gw     = (const float*)d_in[1];
    const float* x_ch     = (const float*)d_in[2];
    const int*   src_swgw = (const int*)d_in[6];
    const int*   dst_swgw = (const int*)d_in[7];
    const int*   src_sw   = (const int*)d_in[10];
    const int*   dst_sw   = (const int*)d_in[11];
    const float* W_swgw   = (const float*)d_in[14];
    const float* b_swgw   = (const float*)d_in[15];
    const float* W_sw     = (const float*)d_in[18];
    const float* b_sw     = (const float*)d_in[19];
    const float* Wih0     = (const float*)d_in[20];
    const float* Whh0     = (const float*)d_in[21];
    const float* bih0     = (const float*)d_in[22];
    const float* bhh0     = (const float*)d_in[23];
    const float* Wih1     = (const float*)d_in[24];
    const float* Whh1     = (const float*)d_in[25];
    const float* bih1     = (const float*)d_in[26];
    const float* bhh1     = (const float*)d_in[27];
    const float* fc1W     = (const float*)d_in[28];
    const float* fc1b     = (const float*)d_in[29];
    const float* fc2W     = (const float*)d_in[30];
    const float* fc2b     = (const float*)d_in[31];

    // Packed B sizes (dwords)
    const int PK_GCN = (D_IN / 32) * (D_H / 16) * 256;   // 16384
    const int PK_IH  = (D_H / 32) * (D_G / 16) * 256;    // 131072

    // Workspace carving (hbuf reused for both GCN GEMMs; xg reused for layer 2)
    char* p = (char*)d_ws;
    size_t off = 0;
    auto carve = [&](size_t bytes) {
        void* q = p + off;
        off += (bytes + 255) & ~(size_t)255;
        return q;
    };
    float*    hbuf    = (float*)   carve((size_t)N_GW  * D_H * 4);   // 153.6 MB
    float*    out_hru = (float*)   carve((size_t)N_HRU * D_H * 4);   //   8.4 MB
    float*    xg      = (float*)   carve((size_t)N_HRU * D_G * 4);   //  33.6 MB
    float*    hseq    = (float*)   carve((size_t)N_HRU * D_H * 4);   //   8.4 MB
    unsigned* dAs     = (unsigned*)carve((size_t)N_GW  * 4);
    unsigned* dAd     = (unsigned*)carve((size_t)N_HRU * 4);
    unsigned* dBs     = (unsigned*)carve((size_t)N_CH  * 4);
    unsigned* dBd     = (unsigned*)carve((size_t)N_HRU * 4);
    unsigned* BpA     = (unsigned*)carve((size_t)PK_GCN * 4);        // W_swgw packed
    unsigned* BpB     = (unsigned*)carve((size_t)PK_GCN * 4);        // W_sw packed
    unsigned* BpI0    = (unsigned*)carve((size_t)PK_IH  * 4);        // Wih0 packed
    unsigned* BpI1    = (unsigned*)carve((size_t)PK_IH  * 4);        // Wih1 packed

    // 0) pack all weight matrices into bf16 WMMA fragment layout
    k_pack<<<(PK_GCN + 255) / 256, 256, 0, stream>>>(W_swgw, BpA, D_IN, D_H, 0);
    k_pack<<<(PK_GCN + 255) / 256, 256, 0, stream>>>(W_sw,   BpB, D_IN, D_H, 0);
    k_pack<<<(PK_IH  + 255) / 256, 256, 0, stream>>>(Wih0,   BpI0, D_H, D_G, 1);
    k_pack<<<(PK_IH  + 255) / 256, 256, 0, stream>>>(Wih1,   BpI1, D_H, D_G, 1);

    // 1) init scratch (must re-run every call: harness does not re-poison)
    k_init<<<(N_HRU * D_H + 255) / 256, 256, 0, stream>>>(
        out_hru, b_swgw, b_sw, dAs, dAd, dBs, dBd);

    // 2) degrees for both live GCNs
    k_deg<<<(E_SWGW + 255) / 256, 256, 0, stream>>>(src_swgw, dst_swgw, dAs, dAd, E_SWGW);
    k_deg<<<(E_SW   + 255) / 256, 256, 0, stream>>>(src_sw,   dst_sw,   dBs, dBd, E_SW);

    // 3) h = x_gw @ W_swgw  (WMMA bf16), then scatter swgw edges into out_hru
    k_gemm<<<dim3(N_GW / 16, D_H / (16 * COLT)), 32, 0, stream>>>(
        x_gw, BpA, nullptr, nullptr, hbuf, D_IN, D_H);
    k_scatter<<<E_SWGW, D_H, 0, stream>>>(hbuf, src_swgw, dst_swgw, dAs, dAd, out_hru);

    // 4) h = x_channel @ W_sw (reuse hbuf), scatter sw edges
    k_gemm<<<dim3(N_CH / 16, D_H / (16 * COLT)), 32, 0, stream>>>(
        x_ch, BpB, nullptr, nullptr, hbuf, D_IN, D_H);
    k_scatter<<<E_SW, D_H, 0, stream>>>(hbuf, src_sw, dst_sw, dBs, dBd, out_hru);

    // 5) layer-1 input contributions: xg = out_hru @ Wih0.T + bih0 + bhh0 (WMMA)
    k_gemm<<<dim3(N_HRU / 16, D_G / (16 * COLT)), 32, 0, stream>>>(
        out_hru, BpI0, bih0, bhh0, xg, D_H, D_G);

    // 6) sequential LSTM layer 1 -> hseq
    k_lstm<<<1, D_G, 0, stream>>>(xg, Whh0, hseq, N_HRU);

    // 7) layer-2 input contributions (xg buffer reused)
    k_gemm<<<dim3(N_HRU / 16, D_G / (16 * COLT)), 32, 0, stream>>>(
        hseq, BpI1, bih1, bhh1, xg, D_H, D_G);

    // 8) sequential LSTM layer 2 -> hseq (reused; only last row consumed)
    k_lstm<<<1, D_G, 0, stream>>>(xg, Whh1, hseq, N_HRU);

    // 9) FC head on h2[-1] -> scalar
    k_head<<<1, 128, 0, stream>>>(hseq + (size_t)(N_HRU - 1) * D_H,
                                  fc1W, fc1b, fc2W, fc2b, (float*)d_out);
}
